// Model_39676907882782
// MI455X (gfx1250) — matrix operand order repaired and checked by reading, not compiled, not run on hardware
//
#include <hip/hip_runtime.h>
#include <hip/hip_bf16.h>
#include <math.h>

// ---------------------------------------------------------------------------
// FlashAttention forward, B=4,H=16,S=2048,D=64 fp32, via v_wmma_f32_16x16x32_f16.
// S^T = K*Q^T (Q pre-scaled by scale*log2e so WMMA output is already in the
// base-2 softmax domain), O^T = V^T*P^T. Softmax reductions are in-lane plus
// one cross-half shuffle. The key->contract-position permutation (swap bits
// 3<->4 per 32-key group) is baked into V^T LDS staging, so P^T B-fragments
// are plain per-lane cvt_pk packs.
// ---------------------------------------------------------------------------

typedef __attribute__((ext_vector_type(16))) _Float16 v16h;
typedef __attribute__((ext_vector_type(8)))  _Float16 v8h;
typedef __attribute__((ext_vector_type(2)))  _Float16 v2h;
typedef __attribute__((ext_vector_type(8)))  float    v8f;
typedef __attribute__((ext_vector_type(4)))  float    f32x4;

#define ATTN_S 2048
#define ATTN_D 64
#define KBLK   64
#define QROWS  128     // 8 waves x 16 query rows

__global__ __launch_bounds__(256)
void fa_fwd_kernel(const float* __restrict__ Qg,
                   const float* __restrict__ Kg,
                   const float* __restrict__ Vg,
                   const float* __restrict__ scale_p,
                   float* __restrict__ Og)
{
    // Strides are multiples of 8 halfs (16B): every v8h LDS access is aligned.
    __shared__ _Float16 sK [KBLK][80];   // [key][d]          10 KB
    __shared__ _Float16 sVt[ATTN_D][72]; // [d][contract-pos]  9 KB

    const int tid  = threadIdx.x;
    const int wave = tid >> 5;
    const int lane = tid & 31;
    const int half = lane >> 4;
    const int l16  = lane & 15;

    const int bh = blockIdx.y;
    const int q0 = blockIdx.x * QROWS + wave * 16;

    const float LOG2E = 1.44269504088896340736f;
    const float sl = scale_p[0] * LOG2E;          // folded into Q below

    const size_t head_off = (size_t)bh * ATTN_S * ATTN_D;
    const float* Qh = Qg + head_off;
    const float* Kh = Kg + head_off;
    const float* Vh = Vg + head_off;
    float*       Oh = Og + head_off;

    // ---- Q^T B-fragments, pre-scaled by scale*log2e (loaded once) ----------
    v16h qb[2];
    {
        const float* qp = Qh + (size_t)(q0 + l16) * ATTN_D;
        #pragma unroll
        for (int s = 0; s < 2; ++s) {
            const int dbase = 32 * s + (half ? 16 : 0);
            f32x4 a0 = *(const f32x4*)(qp + dbase);
            f32x4 a1 = *(const f32x4*)(qp + dbase + 4);
            f32x4 a2 = *(const f32x4*)(qp + dbase + 8);
            f32x4 a3 = *(const f32x4*)(qp + dbase + 12);
            v16h t;
            #pragma unroll
            for (int i = 0; i < 4; ++i) {
                t[i]      = (_Float16)(a0[i] * sl);
                t[4 + i]  = (_Float16)(a1[i] * sl);
                t[8 + i]  = (_Float16)(a2[i] * sl);
                t[12 + i] = (_Float16)(a3[i] * sl);
            }
            qb[s] = t;
        }
    }

    v8f acc[4] = {};                 // O^T tiles: d rows 16t+(j+8*half), col l16
    float mrun = -__builtin_inff();
    float lsum = 0.f;

    // Cooperative staging: each thread owns 2 adjacent key rows x 8 d-cols.
    const int rp  = tid >> 3;           // row pair 0..31 -> keys 2rp, 2rp+1
    const int kr0 = 2 * rp;
    const int dc  = (tid & 7) * 8;      // 0..56
    // contract position: swap bits 3<->4 of key index (preserves bit 0)
    const int vc0 = (kr0 & 39) | ((kr0 & 8) << 1) | ((kr0 & 16) >> 1);

    for (int kb = 0; kb < ATTN_S; kb += KBLK) {
        // ---- stage K (row-major) and V (transposed+permuted, paired) ------
        {
            const float* kp0 = Kh + (size_t)(kb + kr0) * ATTN_D + dc;
            const float* kp1 = kp0 + ATTN_D;
            f32x4 ka0 = *(const f32x4*)kp0;
            f32x4 ka1 = *(const f32x4*)(kp0 + 4);
            f32x4 kb0 = *(const f32x4*)kp1;
            f32x4 kb1 = *(const f32x4*)(kp1 + 4);
            v8h h0, h1;
            #pragma unroll
            for (int i = 0; i < 4; ++i) {
                h0[i] = (_Float16)ka0[i]; h0[4 + i] = (_Float16)ka1[i];
                h1[i] = (_Float16)kb0[i]; h1[4 + i] = (_Float16)kb1[i];
            }
            *(v8h*)&sK[kr0][dc]     = h0;
            *(v8h*)&sK[kr0 + 1][dc] = h1;

            const float* vp0 = Vh + (size_t)(kb + kr0) * ATTN_D + dc;
            const float* vp1 = vp0 + ATTN_D;
            f32x4 va0 = *(const f32x4*)vp0;
            f32x4 va1 = *(const f32x4*)(vp0 + 4);
            f32x4 vb0 = *(const f32x4*)vp1;
            f32x4 vb1 = *(const f32x4*)(vp1 + 4);
            #pragma unroll
            for (int i = 0; i < 4; ++i) {
                v2h p0; p0[0] = (_Float16)va0[i]; p0[1] = (_Float16)vb0[i];
                v2h p1; p1[0] = (_Float16)va1[i]; p1[1] = (_Float16)vb1[i];
                *(v2h*)&sVt[dc + i][vc0]     = p0;
                *(v2h*)&sVt[dc + 4 + i][vc0] = p1;
            }
        }
        __syncthreads();

        if (kb + KBLK < ATTN_S) {
            __builtin_prefetch(Kh + (size_t)(kb + KBLK + kr0) * ATTN_D + dc, 0, 1);
            __builtin_prefetch(Vh + (size_t)(kb + KBLK + kr0) * ATTN_D + dc, 0, 1);
        }

        // ---- S^T = K * Q^T : four 16x16 tiles (keys 16T..16T+15) ----------
        v8f c[4] = {};
        #pragma unroll
        for (int s = 0; s < 2; ++s) {
            const int coff = 32 * s + (half ? 16 : 0);
            #pragma unroll
            for (int T = 0; T < 4; ++T) {
                union { v16h v; v8h h[2]; } ka;
                ka.h[0] = *(const v8h*)&sK[16 * T + l16][coff];
                ka.h[1] = *(const v8h*)&sK[16 * T + l16][coff + 8];
                c[T] = __builtin_amdgcn_wmma_f32_16x16x32_f16(false, ka.v, false, qb[s],
                                                              (short)0, c[T], false, false);
            }
        }

        // ---- online softmax over 64 keys (scores already base-2 domain) ---
        float mx = -__builtin_inff();
        #pragma unroll
        for (int T = 0; T < 4; ++T)
            #pragma unroll
            for (int j = 0; j < 8; ++j)
                mx = fmaxf(mx, c[T][j]);
        mx = fmaxf(mx, __shfl_xor(mx, 16, 32));
        const float mnew  = fmaxf(mrun, mx);
        const float alpha = __builtin_amdgcn_exp2f(mrun - mnew);
        mrun = mnew;

        float ps0 = 0.f, ps1 = 0.f;
        #pragma unroll
        for (int T = 0; T < 4; ++T)
            #pragma unroll
            for (int j = 0; j < 8; ++j) {
                c[T][j] = __builtin_amdgcn_exp2f(c[T][j] - mnew);
                if (T & 1) ps1 += c[T][j]; else ps0 += c[T][j];
            }
        lsum = lsum * alpha + (ps0 + ps1);

        #pragma unroll
        for (int t = 0; t < 4; ++t)
            #pragma unroll
            for (int j = 0; j < 8; ++j)
                acc[t][j] *= alpha;

        // ---- P^T B-fragments: per-lane packing only ------------------------
        v16h pb[2];
        #pragma unroll
        for (int u = 0; u < 2; ++u)
            #pragma unroll
            for (int j = 0; j < 8; ++j) {
                pb[u][j]     = (_Float16)c[2 * u][j];
                pb[u][8 + j] = (_Float16)c[2 * u + 1][j];
            }

        // ---- O^T += V^T * P^T : 4 d-tiles x 2 k-steps ---------------------
        #pragma unroll
        for (int t = 0; t < 4; ++t)
            #pragma unroll
            for (int u = 0; u < 2; ++u) {
                const int coff = 32 * u + (half ? 16 : 0);
                union { v16h v; v8h h[2]; } va;
                va.h[0] = *(const v8h*)&sVt[16 * t + l16][coff];
                va.h[1] = *(const v8h*)&sVt[16 * t + l16][coff + 8];
                acc[t] = __builtin_amdgcn_wmma_f32_16x16x32_f16(false, va.v, false, pb[u],
                                                                (short)0, acc[t], false, false);
            }

        __syncthreads();
    }

    // ---- epilogue ----------------------------------------------------------
    lsum += __shfl_xor(lsum, 16, 32);
    const float inv = __builtin_amdgcn_rcpf(lsum);

    float* orow = Oh + (size_t)(q0 + l16) * ATTN_D + 8 * half;
    #pragma unroll
    for (int t = 0; t < 4; ++t) {
        f32x4 o0, o1;
        #pragma unroll
        for (int j = 0; j < 4; ++j) {
            o0[j] = acc[t][j] * inv;
            o1[j] = acc[t][4 + j] * inv;
        }
        *(f32x4*)(orow + 16 * t)     = o0;
        *(f32x4*)(orow + 16 * t + 4) = o1;
    }
}

extern "C" void kernel_launch(void* const* d_in, const int* in_sizes, int n_in,
                              void* d_out, int out_size, void* d_ws, size_t ws_size,
                              hipStream_t stream) {
    // setup_inputs dict order: k, q, v, scale_factor, dropout_p
    const float* k     = (const float*)d_in[0];
    const float* q     = (const float*)d_in[1];
    const float* v     = (const float*)d_in[2];
    const float* scale = (const float*)d_in[3];
    (void)in_sizes; (void)n_in; (void)d_ws; (void)ws_size; (void)out_size;

    float* out = (float*)d_out;

    dim3 grid(ATTN_S / QROWS, 4 * 16, 1);
    dim3 block(256, 1, 1);
    fa_fwd_kernel<<<grid, block, 0, stream>>>(q, k, v, scale, out);
}
